// GraphSAGE_10892037062819
// MI455X (gfx1250) — matrix operand order repaired and checked by reading, not compiled, not run on hardware
//
#include <hip/hip_runtime.h>
#include <hip/hip_bf16.h>

// ---------------------------------------------------------------------------
// GraphSAGE (3-layer) on MI455X / gfx1250.
//   dims: 128 -> 256 -> 128 -> 64, N = 100000 nodes, E = 600000 edges.
// Aggregation (scatter-mean) is HBM/atomic bound; GEMMs use bf16 WMMA
// (v_wmma_f32_16x16x32_bf16) with f32 accumulation.
// ---------------------------------------------------------------------------

typedef __attribute__((ext_vector_type(16))) __bf16 v16bf;
typedef __attribute__((ext_vector_type(8)))  __bf16 v8bf;
typedef __attribute__((ext_vector_type(8)))  float  v8f;

#define N_NODES 100000

// ---------------------------------------------------------------------------
// degree + inverse-degree
// ---------------------------------------------------------------------------
__global__ void degree_kernel(const long long* __restrict__ dst, float* __restrict__ deg, int E) {
    int e = blockIdx.x * blockDim.x + threadIdx.x;
    if (e < E) atomicAdd(&deg[(int)dst[e]], 1.0f);
}

__global__ void invdeg_kernel(float* __restrict__ deg, int M) {
    int i = blockIdx.x * blockDim.x + threadIdx.x;
    if (i < M) deg[i] = 1.0f / fmaxf(deg[i], 1.0f);
}

// ---------------------------------------------------------------------------
// scatter-add of source rows into agg[dst] (float4 per thread, f32 atomics)
// ---------------------------------------------------------------------------
template <int C>
__global__ __launch_bounds__(256) void scatter_add_kernel(
    const float* __restrict__ feat, const long long* __restrict__ src,
    const long long* __restrict__ dst, float* __restrict__ agg, int E) {
    constexpr int TPE = C / 4;  // threads per edge (power of 2)
    long long gid = (long long)blockIdx.x * blockDim.x + threadIdx.x;
    int e = (int)(gid / TPE);
    int t = (int)(gid % TPE);
    if (e >= E) return;
    int s = (int)src[e];
    int d = (int)dst[e];
    float4 v = *reinterpret_cast<const float4*>(feat + (size_t)s * C + t * 4);
    float* p = agg + (size_t)d * C + t * 4;
    atomicAdd(p + 0, v.x);
    atomicAdd(p + 1, v.y);
    atomicAdd(p + 2, v.z);
    atomicAdd(p + 3, v.w);
}

// ---------------------------------------------------------------------------
// weight convert + transpose: W[K x N] f32 -> Wt[N x K] bf16
// ---------------------------------------------------------------------------
__global__ void w_transpose_kernel(const float* __restrict__ W, __bf16* __restrict__ Wt,
                                   int K, int N) {
    int i = blockIdx.x * blockDim.x + threadIdx.x;
    if (i >= K * N) return;
    int k = i / N, n = i % N;
    Wt[(size_t)n * K + k] = (__bf16)W[i];
}

// ---------------------------------------------------------------------------
// convert 8 consecutive f32 (scaled) into bf16 vector slots [BASE..BASE+7]
// ---------------------------------------------------------------------------
template <int BASE>
__device__ __forceinline__ void cvt8(v16bf& d, const float* __restrict__ p, float s) {
    float4 a = *reinterpret_cast<const float4*>(p);
    float4 b = *reinterpret_cast<const float4*>(p + 4);
    d[BASE + 0] = (__bf16)(a.x * s);
    d[BASE + 1] = (__bf16)(a.y * s);
    d[BASE + 2] = (__bf16)(a.z * s);
    d[BASE + 3] = (__bf16)(a.w * s);
    d[BASE + 4] = (__bf16)(b.x * s);
    d[BASE + 5] = (__bf16)(b.y * s);
    d[BASE + 6] = (__bf16)(b.z * s);
    d[BASE + 7] = (__bf16)(b.w * s);
}

// ---------------------------------------------------------------------------
// Fused SAGE GEMM:  out = relu?( (agg * inv_deg) @ Wl + x @ Wr + b )
// One wave -> one 16x16 output tile. Block = 4 waves = 4 N-tiles.
// grid = (M/16, N/64), block = 128.
// A layout (16x32 bf16): lane<16: K {k..k+7, k+16..k+23}; lane>=16: +8.
// B layout (32x16 bf16): lane holds col (lane&15), same K as A.
// ---------------------------------------------------------------------------
template <bool RELU>
__global__ __launch_bounds__(128) void sage_gemm_kernel(
    const float* __restrict__ xrow, const float* __restrict__ agg,
    const float* __restrict__ invdeg, const __bf16* __restrict__ Wlt,
    const __bf16* __restrict__ Wrt, const float* __restrict__ bias,
    float* __restrict__ out, int K, int N) {
    const int lane   = threadIdx.x & 31;
    const int wave   = threadIdx.x >> 5;
    const int tile_m = blockIdx.x << 4;
    const int tile_n = ((blockIdx.y << 2) + wave) << 4;

    const int mrow = tile_m + (lane & 15);   // row this lane feeds into A
    const int ncol = tile_n + (lane & 15);   // column for B / C / D
    const int aoff = (lane >> 4) << 3;       // 0 or 8  (A K-half select)
    const int boff = (lane >> 4) << 4;       // 0 or 16 (unused)

    const float idg = invdeg[mrow];
    const float bv  = bias[ncol];
    v8f acc = {bv, bv, bv, bv, bv, bv, bv, bv};

    const float*  ap = agg  + (size_t)mrow * K;
    const float*  xp = xrow + (size_t)mrow * K;
    const __bf16* wl = Wlt  + (size_t)ncol * K;
    const __bf16* wr = Wrt  + (size_t)ncol * K;

#pragma unroll 4
    for (int kk = 0; kk < K; kk += 32) {
        v16bf am, ax;
        cvt8<0>(am, ap + kk + aoff,       idg);
        cvt8<8>(am, ap + kk + 16 + aoff,  idg);
        cvt8<0>(ax, xp + kk + aoff,       1.0f);
        cvt8<8>(ax, xp + kk + 16 + aoff,  1.0f);
        v16bf bl = __builtin_shufflevector(*reinterpret_cast<const v8bf*>(wl + kk + aoff), *reinterpret_cast<const v8bf*>(wl + kk + 16 + aoff), 0, 1, 2, 3, 4, 5, 6, 7, 8, 9, 10, 11, 12, 13, 14, 15);
        v16bf br = __builtin_shufflevector(*reinterpret_cast<const v8bf*>(wr + kk + aoff), *reinterpret_cast<const v8bf*>(wr + kk + 16 + aoff), 0, 1, 2, 3, 4, 5, 6, 7, 8, 9, 10, 11, 12, 13, 14, 15);
        acc = __builtin_amdgcn_wmma_f32_16x16x32_bf16(false, am, false, bl,
                                                      (short)0, acc, false, false);
        acc = __builtin_amdgcn_wmma_f32_16x16x32_bf16(false, ax, false, br,
                                                      (short)0, acc, false, false);
    }

    // C/D layout: VGPR r -> row (r + 8*(lane>=16)), col (lane&15)
    const int mbase = tile_m + ((lane >> 4) << 3);
    float* op = out + (size_t)mbase * N + ncol;
#pragma unroll
    for (int r = 0; r < 8; ++r) {
        float v = acc[r];
        if (RELU) v = fmaxf(v, 0.0f);
        op[(size_t)r * N] = v;
    }
}

// ---------------------------------------------------------------------------
// log_softmax over 64 classes, one wave (32 lanes x 2 classes) per node
// ---------------------------------------------------------------------------
__global__ __launch_bounds__(256) void log_softmax64_kernel(float* __restrict__ out, int M) {
    const int lane = threadIdx.x & 31;
    const int node = blockIdx.x * 8 + (threadIdx.x >> 5);
    if (node >= M) return;
    float* row = out + (size_t)node * 64;
    float2 v = *reinterpret_cast<float2*>(row + lane * 2);
    float m = fmaxf(v.x, v.y);
#pragma unroll
    for (int o = 16; o > 0; o >>= 1) m = fmaxf(m, __shfl_xor(m, o, 32));
    float s = expf(v.x - m) + expf(v.y - m);
#pragma unroll
    for (int o = 16; o > 0; o >>= 1) s += __shfl_xor(s, o, 32);
    float lse = m + logf(s);
    float2 r = {v.x - lse, v.y - lse};
    *reinterpret_cast<float2*>(row + lane * 2) = r;
}

// ---------------------------------------------------------------------------
// launcher
// ---------------------------------------------------------------------------
static inline size_t align_up(size_t x) { return (x + 255) & ~(size_t)255; }

extern "C" void kernel_launch(void* const* d_in, const int* in_sizes, int n_in,
                              void* d_out, int out_size, void* d_ws, size_t ws_size,
                              hipStream_t stream) {
    const float*     x   = (const float*)d_in[0];
    const long long* ei  = (const long long*)d_in[1];
    const float*     W1l = (const float*)d_in[2];
    const float*     W1r = (const float*)d_in[3];
    const float*     b1  = (const float*)d_in[4];
    const float*     W2l = (const float*)d_in[5];
    const float*     W2r = (const float*)d_in[6];
    const float*     b2  = (const float*)d_in[7];
    const float*     W3l = (const float*)d_in[8];
    const float*     W3r = (const float*)d_in[9];
    const float*     b3  = (const float*)d_in[10];
    float* out = (float*)d_out;

    const int M = N_NODES;
    const int E = in_sizes[1] / 2;
    const long long* src = ei;
    const long long* dst = ei + E;

    // ---- workspace carve-up -------------------------------------------------
    char*  ws  = (char*)d_ws;
    size_t off = 0;
    float* deg = (float*)(ws + off); off += align_up((size_t)M * 4);                  // deg -> inv_deg
    float* agg = (float*)(ws + off); off += align_up((size_t)M * 256 * 4);            // max-width agg
    float* h1  = (float*)(ws + off); off += align_up((size_t)M * 256 * 4);
    float* h2  = (float*)(ws + off); off += align_up((size_t)M * 128 * 4);
    __bf16* w1lt = (__bf16*)(ws + off); off += align_up((size_t)128 * 256 * 2);
    __bf16* w1rt = (__bf16*)(ws + off); off += align_up((size_t)128 * 256 * 2);
    __bf16* w2lt = (__bf16*)(ws + off); off += align_up((size_t)256 * 128 * 2);
    __bf16* w2rt = (__bf16*)(ws + off); off += align_up((size_t)256 * 128 * 2);
    __bf16* w3lt = (__bf16*)(ws + off); off += align_up((size_t)128 * 64 * 2);
    __bf16* w3rt = (__bf16*)(ws + off); off += align_up((size_t)128 * 64 * 2);

    // ---- degree / inverse degree (edge list is layer-invariant) -------------
    hipMemsetAsync(deg, 0, (size_t)M * 4, stream);
    degree_kernel<<<(E + 255) / 256, 256, 0, stream>>>(dst, deg, E);
    invdeg_kernel<<<(M + 255) / 256, 256, 0, stream>>>(deg, M);

    // ---- weights -> bf16, transposed to [N x K] -----------------------------
    w_transpose_kernel<<<(128 * 256 + 255) / 256, 256, 0, stream>>>(W1l, w1lt, 128, 256);
    w_transpose_kernel<<<(128 * 256 + 255) / 256, 256, 0, stream>>>(W1r, w1rt, 128, 256);
    w_transpose_kernel<<<(256 * 128 + 255) / 256, 256, 0, stream>>>(W2l, w2lt, 256, 128);
    w_transpose_kernel<<<(256 * 128 + 255) / 256, 256, 0, stream>>>(W2r, w2rt, 256, 128);
    w_transpose_kernel<<<(128 * 64 + 255) / 256, 256, 0, stream>>>(W3l, w3lt, 128, 64);
    w_transpose_kernel<<<(128 * 64 + 255) / 256, 256, 0, stream>>>(W3r, w3rt, 128, 64);

    // ---- layer 1: 128 -> 256, ReLU ------------------------------------------
    hipMemsetAsync(agg, 0, (size_t)M * 128 * 4, stream);
    {
        long long total = (long long)E * (128 / 4);
        scatter_add_kernel<128><<<(int)((total + 255) / 256), 256, 0, stream>>>(x, src, dst, agg, E);
    }
    sage_gemm_kernel<true><<<dim3(M / 16, 256 / 64), 128, 0, stream>>>(
        x, agg, deg, w1lt, w1rt, b1, h1, 128, 256);

    // ---- layer 2: 256 -> 128, ReLU ------------------------------------------
    hipMemsetAsync(agg, 0, (size_t)M * 256 * 4, stream);
    {
        long long total = (long long)E * (256 / 4);
        scatter_add_kernel<256><<<(int)((total + 255) / 256), 256, 0, stream>>>(h1, src, dst, agg, E);
    }
    sage_gemm_kernel<true><<<dim3(M / 16, 128 / 64), 128, 0, stream>>>(
        h1, agg, deg, w2lt, w2rt, b2, h2, 256, 128);

    // ---- layer 3: 128 -> 64, no ReLU, straight into d_out -------------------
    hipMemsetAsync(agg, 0, (size_t)M * 128 * 4, stream);
    {
        long long total = (long long)E * (128 / 4);
        scatter_add_kernel<128><<<(int)((total + 255) / 256), 256, 0, stream>>>(h2, src, dst, agg, E);
    }
    sage_gemm_kernel<false><<<dim3(M / 16, 64 / 64), 128, 0, stream>>>(
        h2, agg, deg, w3lt, w3rt, b3, out, 128, 64);

    // ---- log_softmax over 64 classes ----------------------------------------
    log_softmax64_kernel<<<(M + 7) / 8, 256, 0, stream>>>(out, M);
}
